// LongSelfAttention_10746008175577
// MI455X (gfx1250) — matrix operand order repaired and checked by reading, not compiled, not run on hardware
//
#include <hip/hip_runtime.h>

#define BB   2
#define TSEQ 4096
#define EE   768
#define HH   12
#define SD   64
#define WW   256
// padded time axis: 256 left pad, 288 right pad (last key tile reaches q0+287)
#define OFF  256
#define TP   (TSEQ + 256 + 288)   // 4640

typedef __attribute__((ext_vector_type(16))) __bf16 v16bf;
typedef __attribute__((ext_vector_type(8)))  __bf16 v8bf;
typedef __attribute__((ext_vector_type(8)))  float  v8f;

__device__ __forceinline__ v8f wmma_bf16(v16bf a, v16bf b, v8f c) {
  // D = A(16x32) * B(32x16) + C, f32 accumulate
  return __builtin_amdgcn_wmma_f32_16x16x32_bf16(false, a, false, b, (short)0, c, false, false);
}

__global__ void cvt_f32_bf16(const float* __restrict__ src, __bf16* __restrict__ dst, int n) {
  int i = blockIdx.x * blockDim.x + threadIdx.x;
  if (i < n) dst[i] = (__bf16)src[i];
}

__global__ void zero_bf16(__bf16* __restrict__ dst, int n) {
  int i = blockIdx.x * blockDim.x + threadIdx.x;
  if (i < n) dst[i] = (__bf16)0.0f;
}

// One wave computes a 16(t) x 64(e within one head) tile of y = x @ W^T.
// z = 0:q (scaled by 1/8), 1:k  -> layout [b,h,TP,SD] at +OFF
// z = 2:v                       -> layout [b,h,SD,TP] at +OFF (transposed)
__global__ __launch_bounds__(32)
void proj_wmma(const __bf16* __restrict__ xb,   // [B*T, E]
               const __bf16* __restrict__ wb,   // [3][E, E] (row: out-dim, col: in-dim)
               __bf16* __restrict__ qk,         // q then k, each B*H*TP*SD
               __bf16* __restrict__ vt)         // B*H*SD*TP
{
  const int lane = threadIdx.x & 31;
  const int g    = lane >> 4;        // half-wave group
  const int c15  = lane & 15;
  const int row0 = blockIdx.x * 16;  // tile base row in [0, B*T)
  const int h    = blockIdx.y;       // head == 64-col group
  const int z    = blockIdx.z;       // 0=q 1=k 2=v

  const __bf16* W = wb + (size_t)z * EE * EE;
  const int arow = row0 + c15;       // A-fragment row (M = lane&15)

  v8f acc[4] = {};
  for (int kt = 0; kt < EE; kt += 32) {
    // A fragment from x: elems 0..15 = K[g*16..g*16+15]
    const __bf16* ap = xb + (size_t)arow * EE + kt + g * 16;
    v8bf lo = *(const v8bf*)(ap);
    v8bf hi = *(const v8bf*)(ap + 8);
    v16bf a;
    for (int j = 0; j < 8; j++) { a[j] = lo[j]; a[j + 8] = hi[j]; }
    for (int f = 0; f < 4; f++) {
      // B fragment: B[k][n] = W[n][k]; lane col n, 16 contiguous k per lane
      const int col = h * 64 + f * 16 + c15;
      v16bf bv = *(const v16bf*)(W + (size_t)col * EE + kt + g * 16);
      acc[f] = wmma_bf16(a, bv, acc[f]);
    }
  }

  const int b  = row0 / TSEQ;
  const int t0 = row0 % TSEQ;
  const float scale = (z == 0) ? 0.125f : 1.0f;   // 1/sqrt(64)

  if (z < 2) {
    __bf16* dst = qk + (size_t)z * ((size_t)BB * HH * TP * SD)
                     + (((size_t)b * HH + h) * TP + OFF + t0) * SD;
    for (int f = 0; f < 4; f++) {
      const int sd = f * 16 + c15;
      for (int r = 0; r < 8; r++) {
        const int t = g * 8 + r;   // C rows: M = 8g + r
        dst[(size_t)t * SD + sd] = (__bf16)(acc[f][r] * scale);
      }
    }
  } else {
    __bf16* dst = vt + (((size_t)b * HH + h) * SD) * TP;
    for (int f = 0; f < 4; f++) {
      const int sd = f * 16 + c15;
      v8bf pk;
      for (int r = 0; r < 8; r++) pk[r] = (__bf16)acc[f][r];
      *(v8bf*)(dst + (size_t)sd * TP + OFF + t0 + g * 8) = pk;  // t-contiguous
    }
  }
}

// One wave: 16 queries of one (b,h), flash-attention over band [q-256, q+256].
// Computes S^T = K_tile x Q^T so the P fragments are already in A layout for PV.
__global__ __launch_bounds__(32)
void attn_wmma(const __bf16* __restrict__ qb,  // [B,H,TP,SD]
               const __bf16* __restrict__ kb,  // [B,H,TP,SD]
               const __bf16* __restrict__ vt,  // [B,H,SD,TP]
               float* __restrict__ out)        // [B,T,E] fp32
{
  const int lane = threadIdx.x & 31;
  const int g    = lane >> 4;
  const int c15  = lane & 15;
  const int q0   = blockIdx.x * 16;
  const int h    = blockIdx.y;
  const int b    = blockIdx.z;
  const int myq  = q0 + c15;          // this lane's query (B/C col = lane&15)

  const __bf16* qptr = qb + (((size_t)b * HH + h) * TP + OFF + q0) * SD;
  const __bf16* kptr = kb + (((size_t)b * HH + h) * TP) * SD;
  const __bf16* vptr = vt + (((size_t)b * HH + h) * SD) * TP;

  // Q as B-fragments (col = query, 16 contiguous s-dims per lane)
  v16bf qf[2];
  for (int d = 0; d < 2; d++)
    qf[d] = *(const v16bf*)(qptr + (size_t)c15 * SD + d * 32 + g * 16);

  float m = -1e30f, l = 0.0f;
  v8f o[4] = {};

  for (int it = 0; it < 17; it++) {
    const int tb = q0 - WW + it * 32;           // key tile base (may be <0 / >=T)

    // K as A-fragments [keyHalf][dimHalf]
    v16bf ka[2][2];
    for (int kh = 0; kh < 2; kh++) {
      const __bf16* rp = kptr + (size_t)(tb + 16 * kh + c15 + OFF) * SD;
      for (int dh = 0; dh < 2; dh++) {
        v8bf lo = *(const v8bf*)(rp + dh * 32 + g * 16);
        v8bf hi = *(const v8bf*)(rp + dh * 32 + g * 16 + 8);
        v16bf a;
        for (int j = 0; j < 8; j++) { a[j] = lo[j]; a[j + 8] = hi[j]; }
        ka[kh][dh] = a;
      }
    }

    // S^T tiles: rows = keys, cols = queries
    v8f s0 = {}, s1 = {};
    s0 = wmma_bf16(ka[0][0], qf[0], s0);
    s0 = wmma_bf16(ka[0][1], qf[1], s0);
    s1 = wmma_bf16(ka[1][0], qf[0], s1);
    s1 = wmma_bf16(ka[1][1], qf[1], s1);

    // band mask + per-query (per-lane) running max
    float tmax = -1e30f;
    for (int r = 0; r < 8; r++) {
      const int key0 = tb + g * 8 + r;          // S^T rows: M = 8g + r
      const int key1 = key0 + 16;
      const bool ok0 = (key0 >= 0) && (key0 < TSEQ) && (key0 >= myq - WW) && (key0 <= myq + WW);
      const bool ok1 = (key1 >= 0) && (key1 < TSEQ) && (key1 >= myq - WW) && (key1 <= myq + WW);
      s0[r] = ok0 ? s0[r] : -1e30f;
      s1[r] = ok1 ? s1[r] : -1e30f;
      tmax = fmaxf(tmax, fmaxf(s0[r], s1[r]));
    }
    tmax = fmaxf(tmax, __shfl_xor(tmax, 16, 32));

    const float mn    = fmaxf(m, tmax);
    const float alpha = __expf(m - mn);
    m = mn;

    float p0[8], p1[8], psum = 0.0f;
    for (int r = 0; r < 8; r++) {
      p0[r] = (s0[r] > -1e29f) ? __expf(s0[r] - mn) : 0.0f;
      p1[r] = (s1[r] > -1e29f) ? __expf(s1[r] - mn) : 0.0f;
      psum += p0[r] + p1[r];
    }
    psum += __shfl_xor(psum, 16, 32);
    l = l * alpha + psum;

    // rescale O (O rows are queries 8g+r; alpha lives in lane (8g+r))
    float ar[8];
    for (int r = 0; r < 8; r++) ar[r] = __shfl(alpha, g * 8 + r, 32);
    for (int f = 0; f < 4; f++)
      for (int r = 0; r < 8; r++) o[f][r] *= ar[r];

    // P already in A-fragment layout (S^T C-frags == A-frag key mapping)
    v16bf pa;
    for (int r = 0; r < 8; r++) { pa[r] = (__bf16)p0[r]; pa[r + 8] = (__bf16)p1[r]; }

    // V as B-fragments from transposed layout: t = g*8..g*8+7 and g*8+16..g*8+23 per lane
    for (int f = 0; f < 4; f++) {
      const int sd = f * 16 + c15;
      const __bf16* vp = vptr + (size_t)sd * TP + (OFF + tb) + g * 8;
      v8bf vlo = *(const v8bf*)(vp);
      v8bf vhi = *(const v8bf*)(vp + 16);
      v16bf vb;
      for (int j = 0; j < 8; j++) { vb[j] = vlo[j]; vb[j + 8] = vhi[j]; }
      o[f] = wmma_bf16(pa, vb, o[f]);
    }
  }

  const float li = 1.0f / l;
  float lr[8];
  for (int r = 0; r < 8; r++) lr[r] = __shfl(li, g * 8 + r, 32);
  for (int f = 0; f < 4; f++) {
    const int sd = f * 16 + c15;
    for (int r = 0; r < 8; r++) {
      const int t = q0 + g * 8 + r;
      out[((size_t)b * TSEQ + t) * EE + h * SD + sd] = o[f][r] * lr[r];
    }
  }
}

extern "C" void kernel_launch(void* const* d_in, const int* in_sizes, int n_in,
                              void* d_out, int out_size, void* d_ws, size_t ws_size,
                              hipStream_t stream) {
  const float* x  = (const float*)d_in[0];
  const float* Wq = (const float*)d_in[1];
  const float* Wk = (const float*)d_in[2];
  const float* Wv = (const float*)d_in[3];
  float* out = (float*)d_out;

  const size_t nX   = (size_t)BB * TSEQ * EE;          // x elements
  const size_t nW   = (size_t)EE * EE;                 // one weight matrix
  const size_t nQKV = (size_t)BB * HH * TP * SD;       // one padded q/k/vT buffer

  __bf16* xb   = (__bf16*)d_ws;
  __bf16* wb   = xb + nX;            // 3 weight matrices
  __bf16* qbuf = wb + 3 * nW;        // q
  __bf16* kbuf = qbuf + nQKV;        // k
  __bf16* vtb  = kbuf + nQKV;        // v transposed

  cvt_f32_bf16<<<(int)((nX + 255) / 256), 256, 0, stream>>>(x,  xb,          (int)nX);
  cvt_f32_bf16<<<(int)((nW + 255) / 256), 256, 0, stream>>>(Wq, wb,          (int)nW);
  cvt_f32_bf16<<<(int)((nW + 255) / 256), 256, 0, stream>>>(Wk, wb + nW,     (int)nW);
  cvt_f32_bf16<<<(int)((nW + 255) / 256), 256, 0, stream>>>(Wv, wb + 2 * nW, (int)nW);

  // zero q/k/vT (pads must be 0 so out-of-band V reads contribute exactly 0)
  zero_bf16<<<(int)((3 * nQKV + 255) / 256), 256, 0, stream>>>(qbuf, (int)(3 * nQKV));

  dim3 pg(BB * TSEQ / 16, HH, 3);
  proj_wmma<<<pg, 32, 0, stream>>>(xb, wb, qbuf, vtb);

  dim3 ag(TSEQ / 16, HH, BB);
  attn_wmma<<<ag, 32, 0, stream>>>(qbuf, kbuf, vtb, out);
}
